// SelfAttention_20246475833667
// MI455X (gfx1250) — matrix operand order repaired and checked by reading, not compiled, not run on hardware
//
#include <hip/hip_runtime.h>

typedef __attribute__((ext_vector_type(16))) _Float16 v16h;
typedef __attribute__((ext_vector_type(8)))  _Float16 v8h;
typedef __attribute__((ext_vector_type(8)))  float    v8f;

#define BATCH 4
#define CH    256
#define NPIX  4096
#define DPROJ 320   // 32 q + 32 k + 256 v rows of packed weight
#define QKSTRIDE 64 // per-pixel row in Yqk: d 0..31 = q, 32..63 = k

// ---------------------------------------------------------------------------
// helpers
// ---------------------------------------------------------------------------
static __device__ __forceinline__ v16h cat8(v8h lo, v8h hi) {
  return __builtin_shufflevector(lo, hi, 0, 1, 2, 3, 4, 5, 6, 7,
                                         8, 9, 10, 11, 12, 13, 14, 15);
}

static __device__ __forceinline__ v8f wmma_f16(v16h a, v16h b, v8f c) {
  // D = A(16x32 f16) * B(32x16 f16) + C(16x16 f32)
  return __builtin_amdgcn_wmma_f32_16x16x32_f16(false, a, false, b,
                                                (short)0, c, false, false);
}

// ---------------------------------------------------------------------------
// Kernel 0: pack wq/wk/wv into one f16 weight [DPROJ][CH], biases into f32[DPROJ]
// ---------------------------------------------------------------------------
__global__ __launch_bounds__(256) void pack_weights_kernel(
    const float* __restrict__ wq, const float* __restrict__ bq,
    const float* __restrict__ wk, const float* __restrict__ bk,
    const float* __restrict__ wv, const float* __restrict__ bv,
    _Float16* __restrict__ Wh, float* __restrict__ bias) {
  const int idx = blockIdx.x * blockDim.x + threadIdx.x;   // 0 .. 320*256-1
  const int r = idx >> 8;
  const int c = idx & 255;
  float w, b;
  if (r < 32)      { w = wq[r * CH + c];        b = bq[r]; }
  else if (r < 64) { w = wk[(r - 32) * CH + c]; b = bk[r - 32]; }
  else             { w = wv[(r - 64) * CH + c]; b = bv[r - 64]; }
  Wh[idx] = (_Float16)w;
  if (c == 0) bias[r] = b;
}

// ---------------------------------------------------------------------------
// Kernel 1: fused QKV projection via WMMA, x A-fragments loaded ONCE per wave.
// One wave per (b, 16-pixel tile); loops over all 20 output tiles (160 WMMAs).
//   q/k tiles -> Yqk[b][pix][0..63]   (pixel-major, WMMA B/A fragment friendly)
//   v tiles   -> Vt[b][c][pix]        (channel-major: out-GEMM A-frag = 2xB128)
// ---------------------------------------------------------------------------
__global__ __launch_bounds__(32) void proj_kernel(
    const float* __restrict__ x, const _Float16* __restrict__ Wh,
    const float* __restrict__ bias,
    _Float16* __restrict__ Yqk, _Float16* __restrict__ Vt) {
  const int wid = blockIdx.x;          // b * 256 + nt
  const int nt  = wid & (NPIX / 16 - 1);
  const int b   = wid >> 8;
  const int lane = threadIdx.x;
  const int l15 = lane & 15;
  const int h   = lane >> 4;
  const int n0  = nt << 4;

  const float* xb = x + (size_t)b * CH * NPIX;

  // All 8 A fragments (x^T tile: 16 pixels x 256 channels), kept in VGPRs.
  v16h afr[8];
#pragma unroll
  for (int kc = 0; kc < 8; ++kc) {
#pragma unroll
    for (int e = 0; e < 16; ++e) {
      const int k = e + 16 * h;
      afr[kc][e] = (_Float16)xb[(size_t)(kc * 32 + k) * NPIX + n0 + l15];
    }
  }

  for (int dt = 0; dt < DPROJ / 16; ++dt) {
    const int d0 = dt << 4;
    v8f acc = (v8f){0, 0, 0, 0, 0, 0, 0, 0};
#pragma unroll
    for (int kc = 0; kc < 8; ++kc) {
      // B fragment (W^T): lane col = d0+l15, K = kc*32 + 16h + e (contiguous)
      const _Float16* wrow = Wh + (size_t)(d0 + l15) * CH + kc * 32 + 16 * h;
      acc = wmma_f16(afr[kc], cat8(((const v8h*)wrow)[0], ((const v8h*)wrow)[1]),
                     acc);
    }
    const float bb = bias[d0 + l15];
    // D layout: lane col = outdim d0+l15 ; VGPR r -> pixel n0 + r + 8h
    if (d0 < 64) {
#pragma unroll
      for (int r = 0; r < 8; ++r) {
        const int pix = n0 + r + 8 * h;
        Yqk[((size_t)b * NPIX + pix) * QKSTRIDE + d0 + l15] =
            (_Float16)(acc[r] + bb);
      }
    } else {
      const int c = d0 - 64 + l15;
      v8h vv;
#pragma unroll
      for (int r = 0; r < 8; ++r) vv[r] = (_Float16)(acc[r] + bb);
      // 8 consecutive pixels -> single 16B store
      *(v8h*)(Vt + ((size_t)b * CH + c) * NPIX + n0 + 8 * h) = vv;
    }
  }
}

// ---------------------------------------------------------------------------
// Kernel 2: flash attention, 2 waves per block (each wave owns 128 channels),
// one block per (batch, 16-query tile). Keys streamed 32 at a time.
//   S^T = Ktile(16x32) x Q(32x16)           (2 wmma / chunk)
//   out^T[c,i] += Vtile(16x32) x P^T(32x16) (8 wmma / chunk / wave)
// V fragments are two contiguous B128 loads thanks to channel-major Vt.
// ---------------------------------------------------------------------------
__global__ __launch_bounds__(64) void attn_kernel(
    const float* __restrict__ x, const _Float16* __restrict__ Yqk,
    const _Float16* __restrict__ Vt, const float* __restrict__ gamma_p,
    float* __restrict__ out) {
  const int blk  = blockIdx.x;
  const int qt   = blk & (NPIX / 16 - 1);
  const int b    = blk >> 8;
  const int wave = threadIdx.x >> 5;
  const int lane = threadIdx.x & 31;
  const int l15  = lane & 15;
  const int h    = lane >> 4;
  const int i0   = qt << 4;
  const int cbase = wave << 7;   // 0 or 128

  const _Float16* Qb = Yqk + (size_t)b * NPIX * QKSTRIDE;
  const _Float16* Vb = Vt + (size_t)b * CH * NPIX;

  // Q B-fragment: lane holds query row i0+l15, K=d = 8h + e / 16 + 8h + (e-8)
  v16h qfr;
  {
    const _Float16* qrow = Qb + (size_t)(i0 + l15) * QKSTRIDE + 8 * h;
    qfr = cat8(((const v8h*)qrow)[0], ((const v8h*)(qrow + 16))[0]);
  }

  float m_run = -3.0e38f;
  float l_run = 0.0f;
  v8f oacc[8];
#pragma unroll
  for (int t = 0; t < 8; ++t) oacc[t] = (v8f){0, 0, 0, 0, 0, 0, 0, 0};

  for (int j0 = 0; j0 < NPIX; j0 += 32) {
    // ---- energy tiles S^T[j,i] for 32 keys: two WMMAs, K-dim == 32 exactly
    v8f s0, s1;
    {
      const _Float16* krow =
          Qb + (size_t)(j0 + l15) * QKSTRIDE + 32 + 8 * h;
      s0 = wmma_f16(cat8(((const v8h*)krow)[0], ((const v8h*)(krow + 16))[0]),
                    qfr, (v8f){0, 0, 0, 0, 0, 0, 0, 0});
    }
    {
      const _Float16* krow =
          Qb + (size_t)(j0 + 16 + l15) * QKSTRIDE + 32 + 8 * h;
      s1 = wmma_f16(cat8(((const v8h*)krow)[0], ((const v8h*)(krow + 16))[0]),
                    qfr, (v8f){0, 0, 0, 0, 0, 0, 0, 0});
    }
    // prefetch next chunk's K rows into cache (gfx1250 global_prefetch_b8)
    if (j0 + 32 < NPIX) {
      __builtin_prefetch(Qb + (size_t)(j0 + 32 + l15) * QKSTRIDE + 32, 0, 0);
      __builtin_prefetch(Qb + (size_t)(j0 + 48 + l15) * QKSTRIDE + 32, 0, 0);
    }

    // ---- online softmax: every lane's 16 S values belong to query i = l15
    float mloc = s0[0];
#pragma unroll
    for (int r = 1; r < 8; ++r) mloc = fmaxf(mloc, s0[r]);
#pragma unroll
    for (int r = 0; r < 8; ++r) mloc = fmaxf(mloc, s1[r]);
    mloc = fmaxf(mloc, __shfl_xor(mloc, 16, 32));
    const float mnew  = fmaxf(m_run, mloc);
    const float alpha = __expf(m_run - mnew);

    float p0[8], p1[8];
    float sl = 0.0f;
#pragma unroll
    for (int r = 0; r < 8; ++r) { p0[r] = __expf(s0[r] - mnew); sl += p0[r]; }
#pragma unroll
    for (int r = 0; r < 8; ++r) { p1[r] = __expf(s1[r] - mnew); sl += p1[r]; }
    sl += __shfl_xor(sl, 16, 32);
    l_run = l_run * alpha + sl;
    m_run = mnew;

#pragma unroll
    for (int t = 0; t < 8; ++t)
#pragma unroll
      for (int r = 0; r < 8; ++r) oacc[t][r] *= alpha;

    // ---- convert P^T D-tiles into the next WMMA's B-fragment:
    // lane needs K = 8h + e / 16 + 8h + (e-8) ; all held by this lane
    float o0[8], o1[8];
#pragma unroll
    for (int r = 0; r < 8; ++r) {
      o0[r] = __shfl_xor(p0[r], 16, 32);
      o1[r] = __shfl_xor(p1[r], 16, 32);
    }
    v16h pfr;
#pragma unroll
    for (int e = 0; e < 8; ++e) {
      pfr[e]     = (_Float16)p0[e];
      pfr[e + 8] = (_Float16)p1[e];
    }

    // ---- out^T accumulation over this wave's 128 channels (8 tiles)
#pragma unroll
    for (int t = 0; t < 8; ++t) {
      const int c0 = cbase + (t << 4);
      // A fragment: M = channel row c0+l15, K = key; channel-major Vt makes
      // the 16 halfs two contiguous 16B runs.
      const _Float16* vrow = Vb + (size_t)(c0 + l15) * NPIX + j0 + 8 * h;
      v16h vfr = cat8(((const v8h*)vrow)[0], ((const v8h*)(vrow + 16))[0]);
      if (j0 + 32 < NPIX) __builtin_prefetch(vrow + 32, 0, 0);
      oacc[t] = wmma_f16(vfr, pfr, oacc[t]);
    }
  }

  // ---- epilogue: out = gamma * (o / l) + x
  const float g   = gamma_p[0];
  const float inv = 1.0f / l_run;
  const float* xb = x + (size_t)b * CH * NPIX;
  float* ob       = out + (size_t)b * CH * NPIX;
  const int n = i0 + l15;           // D col -> query index
#pragma unroll
  for (int t = 0; t < 8; ++t) {
#pragma unroll
    for (int r = 0; r < 8; ++r) {
      const int c = cbase + (t << 4) + r + 8 * h;   // D row -> channel
      ob[(size_t)c * NPIX + n] = g * (oacc[t][r] * inv) + xb[(size_t)c * NPIX + n];
    }
  }
}

// ---------------------------------------------------------------------------
// launch
// ---------------------------------------------------------------------------
extern "C" void kernel_launch(void* const* d_in, const int* in_sizes, int n_in,
                              void* d_out, int out_size, void* d_ws, size_t ws_size,
                              hipStream_t stream) {
  const float* x  = (const float*)d_in[0];
  const float* wq = (const float*)d_in[1];
  const float* bq = (const float*)d_in[2];
  const float* wk = (const float*)d_in[3];
  const float* bk = (const float*)d_in[4];
  const float* wv = (const float*)d_in[5];
  const float* bv = (const float*)d_in[6];
  const float* gm = (const float*)d_in[7];
  float* out = (float*)d_out;

  char* ws = (char*)d_ws;
  _Float16* Wh  = (_Float16*)ws;                    // 320*256*2    = 163840 B
  float*    bia = (float*)(ws + 163840);            // 320*4        = 1280 B
  _Float16* Yqk = (_Float16*)(ws + 165120);         // 4*4096*64*2  = 2 MB
  _Float16* Vt  = (_Float16*)(ws + 165120 + 2097152); // 4*256*4096*2 = 8 MB

  pack_weights_kernel<<<dim3(DPROJ), dim3(256), 0, stream>>>(wq, bq, wk, bk,
                                                             wv, bv, Wh, bia);
  proj_kernel<<<dim3(BATCH * (NPIX / 16)), dim3(32), 0, stream>>>(x, Wh, bia,
                                                                  Yqk, Vt);
  attn_kernel<<<dim3(BATCH * (NPIX / 16)), dim3(64), 0, stream>>>(x, Yqk, Vt,
                                                                  gm, out);
}
